// HandLSTM_39779987095732
// MI455X (gfx1250) — matrix operand order repaired and checked by reading, not compiled, not run on hardware
//
#include <hip/hip_runtime.h>
#include <hip/hip_bf16.h>
#include <stdint.h>

// ---------------------------------------------------------------------------
// LSTM for MI455X (gfx1250): bf16 WMMA for the input-projection GEMM and the
// per-step recurrent GEMM; async global->LDS (ASYNCcnt) staging for U and xg.
// ---------------------------------------------------------------------------

typedef __attribute__((ext_vector_type(16))) __bf16 v16bf;
typedef __attribute__((ext_vector_type(8)))  float  v8f;

union Frag {
    uint32_t    u[8];
    __uint128_t o[2];
    v16bf       v;
};
static_assert(sizeof(Frag) == 32, "frag must be 8 VGPRs");

__device__ __forceinline__ uint16_t f2bf(float f) {
    uint32_t u = __float_as_uint(f);
    u += 0x7fffu + ((u >> 16) & 1u);   // round-to-nearest-even
    return (uint16_t)(u >> 16);
}

__device__ __forceinline__ v8f zero8() {
    v8f z;
#pragma unroll
    for (int j = 0; j < 8; ++j) z[j] = 0.0f;
    return z;
}

// CDNA5 async global->LDS: LDS address comes from the VDST VGPR; GVS addressing
// (SGPR64 base + VGPR32 offset). Tracked by ASYNCcnt (in-order completion).
__device__ __forceinline__ void async_ld_b128(uint32_t lds_off, uint64_t sbase, uint32_t voff) {
    asm volatile("global_load_async_to_lds_b128 %0, %1, %2"
                 :: "v"(lds_off), "v"(voff), "s"(sbase) : "memory");
}
#define WAIT_ASYNCCNT(n) asm volatile("s_wait_asynccnt " #n ::: "memory")

// low 32 bits of a generic pointer to LDS == wave-relative LDS byte address
__device__ __forceinline__ uint32_t lds_addr32(const void* p) {
    return (uint32_t)(size_t)p;
}

#define B_    32
#define S_    2048
#define I_    256
#define H_    256
#define G4_   1024
#define NROW  (B_ * S_)          // 65536 rows of x / xg

// ---- workspace layout (bytes) ----
#define WS_XG    ((size_t)0)                                  // 65536*1024 f32 = 256 MB
#define WS_XBF   ((size_t)268435456)                          // 65536*256 bf16 = 32 MB
#define WS_WT    (WS_XBF + (size_t)NROW * I_ * 2)             // Wt[1024][256] bf16
#define WS_UT    (WS_WT  + (size_t)G4_ * I_ * 2)              // Ut[1024][256] bf16
#define WS_BIAS  (WS_UT  + (size_t)G4_ * H_ * 2)              // bias f32 [1024]
#define WS_HSH   (WS_BIAS + (size_t)G4_ * 4)                  // h_shared bf16, 2 bufs * 32*256
#define WS_BAR   (WS_HSH + (size_t)2 * B_ * H_ * 2)           // barrier ints

// ---------------------------------------------------------------------------
// Pack kernels
// ---------------------------------------------------------------------------
__global__ void pack_x(const float* __restrict__ x, uint16_t* __restrict__ xbf) {
    int idx = (blockIdx.x * 256 + threadIdx.x) * 4;
    float4 v = *(const float4*)(x + idx);
    uint32_t lo = (uint32_t)f2bf(v.x) | ((uint32_t)f2bf(v.y) << 16);
    uint32_t hi = (uint32_t)f2bf(v.z) | ((uint32_t)f2bf(v.w) << 16);
    uint32_t* p = (uint32_t*)(xbf + idx);
    p[0] = lo; p[1] = hi;
}

__global__ void pack_wu(const float* __restrict__ wf, const float* __restrict__ wi,
                        const float* __restrict__ wc, const float* __restrict__ wo,
                        const float* __restrict__ uf, const float* __restrict__ ui,
                        const float* __restrict__ uc, const float* __restrict__ uo,
                        uint16_t* __restrict__ wt, uint16_t* __restrict__ ut) {
    int idx = blockIdx.x * 256 + threadIdx.x;     // 0 .. 262143
    int col = idx >> 8;                           // packed gate column 0..1023
    int k   = idx & 255;                          // K index
    int gate = col >> 8;                          // 0=f 1=i 2=c 3=o
    int c    = col & 255;
    const float* wp = (gate == 0) ? wf : (gate == 1) ? wi : (gate == 2) ? wc : wo;
    const float* up = (gate == 0) ? uf : (gate == 1) ? ui : (gate == 2) ? uc : uo;
    wt[(size_t)col * 256 + k] = f2bf(wp[k * 256 + c]);
    ut[(size_t)col * 256 + k] = f2bf(up[k * 256 + c]);
}

__global__ void pack_bias_init(const float* __restrict__ bf, const float* __restrict__ bi,
                               const float* __restrict__ bc, const float* __restrict__ bo,
                               float* __restrict__ biasp, int* __restrict__ bar) {
    int i = blockIdx.x * 256 + threadIdx.x;
    if (i < 1024) {
        int gate = i >> 8, c = i & 255;
        const float* bp = (gate == 0) ? bf : (gate == 1) ? bi : (gate == 2) ? bc : bo;
        biasp[i] = bp[c];
    }
    if (i < 16) bar[i] = 0;   // grid-barrier counter reset (deterministic per launch)
}

// ---------------------------------------------------------------------------
// xg = xbf @ Wt^T + bias. One wave = 16x64 strip, K=256 in 8 WMMA k-steps,
// register double-buffered so loads for kk+1 overlap the 4 WMMAs of kk.
// ---------------------------------------------------------------------------
__global__ void gemm_xw(const uint16_t* __restrict__ xbf, const uint16_t* __restrict__ wt,
                        const float* __restrict__ biasp, float* __restrict__ xg) {
    int tid  = threadIdx.x;
    int wave = tid >> 5, lane = tid & 31;
    int ln   = lane & 15, lg = lane >> 4;
    int gw   = blockIdx.x * 8 + wave;          // 65536 waves total
    int nt   = gw & 15,  mt = gw >> 4;
    int r0   = mt * 16,  c0 = nt * 64;

    v8f acc[4];
#pragma unroll
    for (int n4 = 0; n4 < 4; ++n4) acc[n4] = zero8();

    const char* arow = (const char*)xbf + (size_t)(r0 + ln) * 512;

    auto loadA = [&](int kk, Frag& a) {
        const char* ap = arow + kk * 64 + lg * 16;   // ISA 16-bit A layout: 2x16B
        a.o[0] = *(const __uint128_t*)ap;
        a.o[1] = *(const __uint128_t*)(ap + 32);
    };
    auto loadB = [&](int kk, int n4, Frag& b) {
        int col = c0 + n4 * 16 + ln;
        const char* bp = (const char*)wt + (size_t)col * 512 + kk * 64 + lg * 16;
        b.o[0] = *(const __uint128_t*)bp;
        b.o[1] = *(const __uint128_t*)(bp + 32);
    };

    Frag a_cur, a_nxt, b_cur[4], b_nxt[4];
    loadA(0, a_cur);
#pragma unroll
    for (int n4 = 0; n4 < 4; ++n4) loadB(0, n4, b_cur[n4]);

#pragma unroll
    for (int kk = 0; kk < 8; ++kk) {
        if (kk < 7) {
            loadA(kk + 1, a_nxt);
#pragma unroll
            for (int n4 = 0; n4 < 4; ++n4) loadB(kk + 1, n4, b_nxt[n4]);
        }
#pragma unroll
        for (int n4 = 0; n4 < 4; ++n4)
            acc[n4] = __builtin_amdgcn_wmma_f32_16x16x32_bf16(
                false, a_cur.v, false, b_cur[n4].v, (short)0, acc[n4], false, false);
        if (kk < 7) {
            a_cur = a_nxt;
#pragma unroll
            for (int n4 = 0; n4 < 4; ++n4) b_cur[n4] = b_nxt[n4];
        }
    }
#pragma unroll
    for (int n4 = 0; n4 < 4; ++n4) {
        int col = c0 + n4 * 16 + ln;
        float bv = biasp[col];
#pragma unroll
        for (int j = 0; j < 8; ++j) {
            int m = j + 8 * lg;                       // C/D f32 layout
            xg[(size_t)(r0 + m) * 1024 + col] = acc[n4][j] + bv;
        }
    }
}

// ---------------------------------------------------------------------------
// Recurrent kernel: 4 persistent WGs (one per WGP), each owns 64 hidden units.
// U slice (128 KB bf16) async-loaded to LDS once; per-step xg slice (32 KB)
// double-buffered in LDS via global_load_async_to_lds_b128 issued one full
// step ahead; c-state in LDS; h exchanged via an L2-resident bf16 buffer with
// a monotonic atomic grid barrier.
// ---------------------------------------------------------------------------
#define USTR 528   // 256*2 + 16 pad
#define HSTR 528   // 256*2 + 16 pad
#define ZSTR 1040  // 256*4 + 16 pad
#define XGL_BYTES (2 * 32 * 256 * 4)                 // 2 bufs * [32][256] f32 = 64 KB
#define SMEM_BYTES (256 * USTR + 32 * HSTR + 32 * ZSTR + 32 * 64 * 4 + XGL_BYTES) // 259072

__global__ void lstm_rec(const float* __restrict__ xg, const uint16_t* __restrict__ ut,
                         float* __restrict__ out, float* __restrict__ ht_out,
                         float* __restrict__ ct_out, uint16_t* __restrict__ hsh,
                         int* __restrict__ bar) {
    extern __shared__ char smem[];
    char*  u_lds  = smem;                               // [256 local cols][USTR]
    char*  h_lds  = smem + 256 * USTR;                  // [32 rows][HSTR]   bf16 h
    char*  z_lds  = h_lds + 32 * HSTR;                  // [32 rows][ZSTR]   f32 z
    float* c_lds  = (float*)(z_lds + 32 * ZSTR);        // [32][64] f32 cell state
    char*  xg_lds = (char*)c_lds + 32 * 64 * 4;         // [2][32][256] f32 staged xg

    const int tid  = threadIdx.x;
    const int blk  = blockIdx.x;                        // 0..3 -> hidden chunk of 64
    const int wave = tid >> 5, lane = tid & 31;
    const int ln   = lane & 15, lg = lane >> 4;
    const int n0   = wave * 32;                         // 32 local cols per wave

    const uint32_t u_off   = lds_addr32(u_lds);
    const uint32_t xgl_off = lds_addr32(xg_lds);
    const uint64_t ut_base = (uint64_t)(const void*)ut;
    const uint64_t xg_base = (uint64_t)(const void*)xg;

    // 8 async b128 per thread: stage xg row t -> LDS buffer buf as [b][gate*64+hx]
    auto issue_xg = [&](int tt, int buf) {
#pragma unroll
        for (int i = 0; i < 8; ++i) {
            int q = tid + i * 256;                      // 0..2047
            int b = q >> 6, seg = q & 63;
            int gate = seg >> 4, pos = seg & 15;
            uint32_t voff = (uint32_t)(b * 2048 + tt) * 4096u
                          + (uint32_t)(gate * 256 + blk * 64) * 4u + (uint32_t)pos * 16u;
            uint32_t lds  = xgl_off + (uint32_t)buf * 32768u
                          + (uint32_t)b * 1024u + (uint32_t)gate * 256u + (uint32_t)pos * 16u;
            async_ld_b128(lds, xg_base, voff);
        }
    };

    // ---- async preload of U slice: local col lc = gate*64+hx <-> packed gate*256+blk*64+hx
    for (int q = tid; q < 256 * 32; q += 256) {
        int lc = q >> 5, pos = q & 31;
        int gate = lc >> 6, hx = lc & 63;
        int p = gate * 256 + blk * 64 + hx;
        async_ld_b128(u_off + (uint32_t)lc * USTR + (uint32_t)pos * 16,
                      ut_base, (uint32_t)(p * 512 + pos * 16));
    }
    issue_xg(0, 0);                                     // prologue: stage xg for t=0
    for (int q = tid; q < 32 * HSTR / 4; q += 256) ((uint32_t*)h_lds)[q] = 0u;   // h0 = 0
    for (int q = tid; q < 2048; q += 256) c_lds[q] = 0.0f;                       // c0 = 0
    WAIT_ASYNCCNT(0);
    __syncthreads();

    for (int t = 0; t < S_; ++t) {
        // issue async stage of xg for step t+1 (lands while we do WMMA below)
        issue_xg((t + 1) & (S_ - 1), (t + 1) & 1);

        // ---- phase 1: z = h(t-1) @ U_slice (bf16 WMMA, f32 accumulate) ----
        v8f acc[2][2];
#pragma unroll
        for (int mi = 0; mi < 2; ++mi)
#pragma unroll
            for (int ni = 0; ni < 2; ++ni) acc[mi][ni] = zero8();

#pragma unroll
        for (int kk = 0; kk < 8; ++kk) {
            Frag a[2];
#pragma unroll
            for (int mi = 0; mi < 2; ++mi) {
                const char* ap = h_lds + (size_t)(mi * 16 + ln) * HSTR + kk * 64 + lg * 16;
                a[mi].o[0] = *(const __uint128_t*)ap;
                a[mi].o[1] = *(const __uint128_t*)(ap + 32);
            }
#pragma unroll
            for (int ni = 0; ni < 2; ++ni) {
                int col = n0 + ni * 16 + ln;
                const char* bp = u_lds + (size_t)col * USTR + kk * 64 + lg * 16;
                Frag b;
                b.o[0] = *(const __uint128_t*)bp;
                b.o[1] = *(const __uint128_t*)(bp + 32);
                acc[0][ni] = __builtin_amdgcn_wmma_f32_16x16x32_bf16(
                    false, a[0].v, false, b.v, (short)0, acc[0][ni], false, false);
                acc[1][ni] = __builtin_amdgcn_wmma_f32_16x16x32_bf16(
                    false, a[1].v, false, b.v, (short)0, acc[1][ni], false, false);
            }
        }
#pragma unroll
        for (int mi = 0; mi < 2; ++mi)
#pragma unroll
            for (int ni = 0; ni < 2; ++ni)
#pragma unroll
                for (int j = 0; j < 8; ++j) {
                    int m   = mi * 16 + j + 8 * lg;
                    int col = n0 + ni * 16 + ln;
                    *(float*)(z_lds + (size_t)m * ZSTR + col * 4) = acc[mi][ni][j];
                }
        // wait: <=8 outstanding asyncs (the t+1 batch); in-order completion
        // guarantees the step-t batch has fully landed in LDS.
        WAIT_ASYNCCNT(8);
        __syncthreads();

        // ---- phase 2: gates + state update (256 thr x 8 items) ----
        const float* xgb = (const float*)(xg_lds + (size_t)(t & 1) * 32768);
        for (int q = tid; q < 2048; q += 256) {
            int b  = q >> 6, hx = q & 63;
            const float* zr = (const float*)(z_lds + (size_t)b * ZSTR);
            const float* xr = xgb + b * 256;
            float zf = zr[hx]       + xr[hx];
            float zi = zr[64 + hx]  + xr[64 + hx];
            float zg = zr[128 + hx] + xr[128 + hx];
            float zo = zr[192 + hx] + xr[192 + hx];
            float fg = 1.0f / (1.0f + __expf(-zf));
            float ig = 1.0f / (1.0f + __expf(-zi));
            float eg = __expf(-2.0f * zg);
            float gg = (1.0f - eg) / (1.0f + eg);
            float og = 1.0f / (1.0f + __expf(-zo));
            float c  = fg * c_lds[q] + ig * gg;
            float ec = __expf(-2.0f * c);
            float h  = og * (1.0f - ec) / (1.0f + ec);
            c_lds[q] = c;
            int col = blk * 64 + hx;
            out[((size_t)b * 2048 + t) * 256 + col] = h;
            hsh[(size_t)(t & 1) * (B_ * H_) + b * 256 + col] = f2bf(h);
            if (t == S_ - 1) {
                ht_out[b * 256 + col] = h;
                ct_out[b * 256 + col] = c;
            }
        }
        __threadfence();
        __syncthreads();

        // ---- phase 3: grid barrier across the 4 WGPs (monotonic counter) ----
        if (tid == 0) {
            __hip_atomic_fetch_add(bar, 1, __ATOMIC_RELEASE, __HIP_MEMORY_SCOPE_AGENT);
            int target = 4 * (t + 1);
            while (__hip_atomic_load(bar, __ATOMIC_ACQUIRE, __HIP_MEMORY_SCOPE_AGENT) < target)
                __builtin_amdgcn_s_sleep(1);
        }
        __syncthreads();

        // ---- phase 4: gather full h(t) (all 256 cols, bf16) into LDS ----
        const char* hs = (const char*)(hsh + (size_t)(t & 1) * (B_ * H_));
        for (int q = tid; q < 1024; q += 256) {
            int row = q >> 5, c16 = q & 31;
            *(__uint128_t*)(h_lds + (size_t)row * HSTR + c16 * 16) =
                *(const __uint128_t*)(hs + (size_t)row * 512 + c16 * 16);
        }
        __syncthreads();
    }
}

// ---------------------------------------------------------------------------
extern "C" void kernel_launch(void* const* d_in, const int* in_sizes, int n_in,
                              void* d_out, int out_size, void* d_ws, size_t ws_size,
                              hipStream_t stream) {
    (void)in_sizes; (void)n_in; (void)out_size; (void)ws_size;
    const float* x  = (const float*)d_in[0];
    const float* Wf = (const float*)d_in[1];
    const float* Uf = (const float*)d_in[2];
    const float* bf = (const float*)d_in[3];
    const float* Wi = (const float*)d_in[4];
    const float* Ui = (const float*)d_in[5];
    const float* bi = (const float*)d_in[6];
    const float* Wo = (const float*)d_in[7];
    const float* Uo = (const float*)d_in[8];
    const float* bo = (const float*)d_in[9];
    const float* Wc = (const float*)d_in[10];
    const float* Uc = (const float*)d_in[11];
    const float* bc = (const float*)d_in[12];

    char* ws = (char*)d_ws;
    float*    xg    = (float*)(ws + WS_XG);
    uint16_t* xbf   = (uint16_t*)(ws + WS_XBF);
    uint16_t* wt    = (uint16_t*)(ws + WS_WT);
    uint16_t* utp   = (uint16_t*)(ws + WS_UT);
    float*    biasp = (float*)(ws + WS_BIAS);
    uint16_t* hsh   = (uint16_t*)(ws + WS_HSH);
    int*      bar   = (int*)(ws + WS_BAR);

    float* out = (float*)d_out;                       // hidden_seq [32][2048][256]
    float* ht  = out + (size_t)B_ * S_ * H_;          // h_t [32][256]
    float* ct  = ht + (size_t)B_ * H_;                // c_t [32][256]

    pack_x<<<NROW * I_ / 1024, 256, 0, stream>>>(x, xbf);
    pack_wu<<<1024, 256, 0, stream>>>(Wf, Wi, Wc, Wo, Uf, Ui, Uc, Uo, wt, utp);
    pack_bias_init<<<4, 256, 0, stream>>>(bf, bi, bc, bo, biasp, bar);
    gemm_xw<<<8192, 256, 0, stream>>>(xbf, wt, biasp, xg);
    lstm_rec<<<4, 256, SMEM_BYTES, stream>>>(xg, utp, out, ht, ct, hsh, bar);
}
